// AWPINN_74371653697957
// MI455X (gfx1250) — compile-verified, matrix operand order checked by reading, not run on hardware
//
#include <hip/hip_runtime.h>

typedef __attribute__((ext_vector_type(16))) __bf16 v16bf;
typedef __attribute__((ext_vector_type(8)))  float  v8f;

#define KMAX 512

__global__ __launch_bounds__(256) void AWPINN_wavelet_wmma(
    const float* __restrict__ x, const float* __restrict__ y, const float* __restrict__ z,
    const float* __restrict__ wx, const float* __restrict__ bx,
    const float* __restrict__ wy, const float* __restrict__ by,
    const float* __restrict__ wz, const float* __restrict__ bz,
    const float* __restrict__ coeff, const float* __restrict__ bias_p,
    float* __restrict__ out, int N, int K)
{
    // Per-k parameters, packed for wide LDS loads:
    // spA = (wx, bx, wy, by), spB = (wz, bz, wx^2, wy^2), spC = (wz^2, u = scale*coeff)
    __shared__ float4 spA[KMAX];
    __shared__ float4 spB[KMAX];
    __shared__ float2 spC[KMAX];

    const int tid = threadIdx.x;
    for (int k = tid; k < K; k += blockDim.x) {
        float wxk = wx[k], bxk = bx[k];
        float wyk = wy[k], byk = by[k];
        float wzk = wz[k], bzk = bz[k];
        float prod = wxk * wyk * wzk;
        float u = sqrtf(fmaxf(prod, 1e-12f)) * coeff[k];
        spA[k] = make_float4(wxk, bxk, wyk, byk);
        spB[k] = make_float4(wzk, bzk, wxk * wxk, wyk * wyk);
        spC[k] = make_float2(wzk * wzk, u);
    }
    __syncthreads();

    const int lane = tid & 31;
    const int wave = tid >> 5;
    const int j    = lane & 15;   // B/C column; 0..3 map to (out, d2x, d2y, d2z)
    const int hi   = lane >> 4;   // lane half selects which features / which C rows
    const int tile = blockIdx.x * (blockDim.x >> 5) + wave;
    if (tile * 16 >= N) return;   // uniform per-wave; EXEC stays all-1s for WMMA

    const int m = tile * 16 + j;  // this lane's point (rows duplicated across halves per A layout)
    const float px = x[m], py = y[m], pz = z[m];
    const float bias = bias_p[0];

    // Constant block-diagonal selector: B[K, j] = (j == (K & 3)) ? 1 : 0.
    // (B 32x16 layout: VGPR v holds K = 8*hi + 2v, 2v+1 (+8 if v >= 4) for column j = lane&15,
    //  and (8*hi + e + 8*(e >= 8)) & 3 == e & 3, so the pattern is lane-half independent.)
    v16bf bmat;
#pragma unroll
    for (int e = 0; e < 16; ++e)
        bmat[e] = (j == (e & 3)) ? (__bf16)1.0f : (__bf16)0.0f;

    v8f acc = {0.f, 0.f, 0.f, 0.f, 0.f, 0.f, 0.f, 0.f};

    const float kNegHalfLog2e = -0.7213475204444817f;  // exp(-0.5 t^2) = exp2(t^2 * this)
    const int fofs[4] = {0, 1, 4, 5};

    for (int kb = 0; kb < K; kb += 8) {
        // A-layout (16-bit 16x32): lo lanes own K slots {0..7,16..23} -> features kb+{0,1,4,5};
        // hi lanes own K slots {8..15,24..31} -> features kb+{2,3,6,7}.
        const int base = kb + 2 * hi;
        v16bf amat;
#pragma unroll
        for (int t = 0; t < 4; ++t) {
            const int k = base + fofs[t];
            float4 pa = spA[k];
            float4 pb = spB[k];
            float2 pc = spC[k];

            float xt = fmaf(px, pa.x, -pa.y);
            float yt = fmaf(py, pa.z, -pa.w);
            float zt = fmaf(pz, pb.x, -pb.y);
            float xt2 = xt * xt, yt2 = yt * yt, zt2 = zt * zt;
            float ex = __builtin_amdgcn_exp2f(xt2 * kNegHalfLog2e);
            float ey = __builtin_amdgcn_exp2f(yt2 * kNegHalfLog2e);
            float ez = __builtin_amdgcn_exp2f(zt2 * kNegHalfLog2e);
            float xw = -xt * ex, yw = -yt * ey, zw = -zt * ez;
            float gx = pb.z * xt * (3.0f - xt2) * ex;   // d2xw
            float gy = pb.w * yt * (3.0f - yt2) * ey;
            float gz = pc.x * zt * (3.0f - zt2) * ez;
            float u  = pc.y;                            // scale_k * coeff_k (f32-exact)

            float p = u * yw * zw;
            float q = u * xw;
            // amat element 4t+i = u * a_i for this feature (bf16 round happens here only)
            amat[4 * t + 0] = (__bf16)(xw * p);         // u*xw*yw*zw  -> output
            amat[4 * t + 1] = (__bf16)(gx * p);         // u*gx*yw*zw  -> d2u/dx2
            amat[4 * t + 2] = (__bf16)(q * (gy * zw));  // u*xw*gy*zw  -> d2u/dy2
            amat[4 * t + 3] = (__bf16)(q * (yw * gz));  // u*xw*yw*gz  -> d2u/dz2
        }
        acc = __builtin_amdgcn_wmma_f32_16x16x32_bf16(
            false, amat, false, bmat, (short)0, acc, false, false);
    }

    // C layout: VGPR r holds (M = r + 8*hi, N = j). Columns 0..3 are the 4 outputs.
    if (j < 4) {
        float* o = out + (size_t)j * (size_t)N;
        const int row = tile * 16 + hi * 8;
        const float add = (j == 0) ? bias : 0.0f;
#pragma unroll
        for (int r = 0; r < 8; ++r)
            o[row + r] = acc[r] + add;
    }
}

extern "C" void kernel_launch(void* const* d_in, const int* in_sizes, int n_in,
                              void* d_out, int out_size, void* d_ws, size_t ws_size,
                              hipStream_t stream) {
    const float* x     = (const float*)d_in[0];
    const float* y     = (const float*)d_in[1];
    const float* z     = (const float*)d_in[2];
    const float* wx    = (const float*)d_in[3];
    const float* bx    = (const float*)d_in[4];
    const float* wy    = (const float*)d_in[5];
    const float* by    = (const float*)d_in[6];
    const float* wz    = (const float*)d_in[7];
    const float* bz    = (const float*)d_in[8];
    const float* coeff = (const float*)d_in[9];
    const float* bias  = (const float*)d_in[10];

    const int N = in_sizes[0];   // 65536
    const int K = in_sizes[3];   // 512

    // 16 points per wave-tile, 8 waves (tiles) per 256-thread block
    const int blocks = N / (16 * 8);
    AWPINN_wavelet_wmma<<<blocks, 256, 0, stream>>>(
        x, y, z, wx, bx, wy, by, wz, bz, coeff, bias, (float*)d_out, N, K);
}
